// PointNet2Seg_SPFE_WSLFA_11123965297225
// MI455X (gfx1250) — matrix operand order repaired and checked by reading, not compiled, not run on hardware
//
#include <hip/hip_runtime.h>
#include <math.h>

typedef __attribute__((ext_vector_type(16))) _Float16 v16h;
typedef __attribute__((ext_vector_type(8)))  _Float16 v8h;
typedef __attribute__((ext_vector_type(8)))  float    v8f;
typedef __attribute__((ext_vector_type(4)))  float    v4f;

union H16 { v16h v; v8h p[2]; _Float16 h[16]; };
union H8  { v8h  v; _Float16 h[8]; };
union F4  { v4f  v; float f[4]; };

__device__ __forceinline__ float red_add16(float v) {
#pragma unroll
  for (int m = 1; m < 16; m <<= 1) v += __shfl_xor(v, m, 32);
  return v;
}
__device__ __forceinline__ float red_max16(float v) {
#pragma unroll
  for (int m = 1; m < 16; m <<= 1) v = fmaxf(v, __shfl_xor(v, m, 32));
  return v;
}

enum { EPI_BNRELU = 0, EPI_FPRIME = 1, EPI_ALPHA = 2, EPI_BIAS = 3 };

// D[o,j] = sum_c Wp[o,c] * act[j,c]   (act point-major, K = Cpad mult of 32)
// Wave computes 16(o) x 32(j); j0 is 32-aligned so a wave owns a full k=32 group.
template<int EPI>
__global__ __launch_bounds__(128)
void gemm_wmma_k(const _Float16* __restrict__ Wp,
                 const float* __restrict__ scl,
                 const float* __restrict__ bia,
                 const _Float16* __restrict__ act, long long actStride,
                 float* __restrict__ outF, long long outFStride,
                 _Float16* __restrict__ outH, long long outHStride,
                 const _Float16* __restrict__ fprIn,
                 _Float16* __restrict__ fprOut,
                 _Float16* __restrict__ alphaOut,
                 int CpadA, int CcatAl,
                 int Nj, int Cpad, int Cout, int Oreal)
{
  const int lane = threadIdx.x & 31;
  const int wave = threadIdx.x >> 5;
  const int r    = lane & 15;
  const int half = lane >> 4;
  const int b    = blockIdx.z;
  const int j0   = blockIdx.x * 128 + wave * 32;
  const int o0   = blockIdx.y * 16;

  const _Float16* actB = act + (size_t)b * actStride;
  // A (16x32 f16): lane<16 row=r holds K kb+0..7 and kb+16..23; lane>=16 row=r holds K kb+8..15, kb+24..31
  const _Float16* ap  = Wp   + (size_t)(o0 + r) * Cpad + half * 8;
  // B (32x16 f16): lane<16 col=r holds K kb+0..7 and kb+16..23; lane>=16 col=r holds K kb+8..15, kb+24..31
  const _Float16* b0p = actB + (size_t)(j0 + r) * Cpad + half * 8;
  const _Float16* b1p = actB + (size_t)(j0 + 16 + r) * Cpad + half * 8;

  v8f c0 = {}; v8f c1 = {};
  for (int kb = 0; kb < Cpad; kb += 32) {
    H16 a, x0, x1;
    a.p[0]  = *(const v8h*)(ap  + kb);
    a.p[1]  = *(const v8h*)(ap  + kb + 16);
    x0.p[0] = *(const v8h*)(b0p + kb);
    x0.p[1] = *(const v8h*)(b0p + kb + 16);
    x1.p[0] = *(const v8h*)(b1p + kb);
    x1.p[1] = *(const v8h*)(b1p + kb + 16);
    c0 = __builtin_amdgcn_wmma_f32_16x16x32_f16(false, a.v, false, x0.v, (short)0, c0, false, false);
    c1 = __builtin_amdgcn_wmma_f32_16x16x32_f16(false, a.v, false, x1.v, (short)0, c1, false, false);
  }

  // D: lane<16 holds rows o0+0..7 for col j0+r (tile0) / j0+16+r (tile1); lane>=16 rows o0+8..15.
  const int rowBase = o0 + half * 8;
  const int jA = j0 + r;
  const int jB = j0 + 16 + r;

  if (EPI == EPI_BIAS) {
    float* oF = outF + (size_t)b * outFStride;   // [Oreal][Nj] channel-major final output
#pragma unroll
    for (int q = 0; q < 8; q++) {
      int ro = rowBase + q;
      if (ro < Oreal) {
        float bb = bia[ro];
        oF[(size_t)ro * Nj + jA] = c0[q] + bb;
        oF[(size_t)ro * Nj + jB] = c1[q] + bb;
      }
    }
    return;
  }

  float v0[8], v1[8];
#pragma unroll
  for (int q = 0; q < 8; q++) {
    float s = scl[rowBase + q], t = bia[rowBase + q];
    v0[q] = fmaxf(fmaf(c0[q], s, t), 0.f);
    v1[q] = fmaxf(fmaf(c1[q], s, t), 0.f);
  }

  if (EPI == EPI_BNRELU) {
    if (outF) {
      float* oF = outF + (size_t)b * outFStride;
      F4 a0, a1, e0, e1;
#pragma unroll
      for (int q = 0; q < 4; q++) { a0.f[q] = v0[q]; a1.f[q] = v0[q + 4]; e0.f[q] = v1[q]; e1.f[q] = v1[q + 4]; }
      *(v4f*)(oF + (size_t)jA * Cout + rowBase)     = a0.v;
      *(v4f*)(oF + (size_t)jA * Cout + rowBase + 4) = a1.v;
      *(v4f*)(oF + (size_t)jB * Cout + rowBase)     = e0.v;
      *(v4f*)(oF + (size_t)jB * Cout + rowBase + 4) = e1.v;
    }
    if (outH) {
      _Float16* oH = outH + (size_t)b * outHStride;
      H8 h0, h1;
#pragma unroll
      for (int q = 0; q < 8; q++) { h0.h[q] = (_Float16)v0[q]; h1.h[q] = (_Float16)v1[q]; }
      *(v8h*)(oH + (size_t)jA * Cout + rowBase) = h0.v;
      *(v8h*)(oH + (size_t)jB * Cout + rowBase) = h1.v;
    }
    return;
  }

  if (EPI == EPI_FPRIME) {
    // mean over k=32 per (row, center): reduce across 16 lanes + tile pair
    H8 p0, p1, m0, m1;
#pragma unroll
    for (int q = 0; q < 8; q++) {
      float mean = red_add16(v0[q] + v1[q]) * (1.0f / 32.0f);
      p0.h[q] = (_Float16)v0[q];
      p1.h[q] = (_Float16)v1[q];
      m0.h[q] = (_Float16)(v0[q] - mean);
      m1.h[q] = (_Float16)(v1[q] - mean);
    }
    *(v8h*)(fprOut + (size_t)jA * Cout + rowBase) = p0.v;
    *(v8h*)(fprOut + (size_t)jB * Cout + rowBase) = p1.v;
    *(v8h*)(alphaOut + (size_t)jA * CpadA + CcatAl + rowBase) = m0.v;
    *(v8h*)(alphaOut + (size_t)jB * CpadA + CcatAl + rowBase) = m1.v;
    return;
  }

  if (EPI == EPI_ALPHA) {
    // softmax over k=32 (per row) then weighted sum against f'
    H8 f0h, f1h;
    f0h.v = *(const v8h*)(fprIn + (size_t)jA * Cout + rowBase);
    f1h.v = *(const v8h*)(fprIn + (size_t)jB * Cout + rowBase);
    float tot[8];
#pragma unroll
    for (int q = 0; q < 8; q++) {
      float mx = red_max16(fmaxf(v0[q], v1[q]));
      float e0 = __expf(v0[q] - mx);
      float e1 = __expf(v1[q] - mx);
      float s  = red_add16(e0 + e1);
      float p  = red_add16(e0 * (float)f0h.h[q] + e1 * (float)f1h.h[q]);
      tot[q] = p / s;
    }
    if (r == 0) {
      float* oF = outF + (size_t)b * outFStride;
      int m = j0 >> 5;
#pragma unroll
      for (int q = 0; q < 8; q++) oF[(size_t)m * Cout + rowBase + q] = tot[q];
    }
    return;
  }
}

// ---- weight pack: f32 [O][C] -> f16 [Opad][Cpad], optional channel remap for alpha concat
__global__ void pack_k(const float* __restrict__ W, const float* __restrict__ bcv,
                       const float* __restrict__ g, const float* __restrict__ bt,
                       _Float16* __restrict__ Wp, float* __restrict__ scl, float* __restrict__ bia,
                       int O, int C, int Opad, int Cpad, int split, int splitAl)
{
  int t = blockIdx.x * blockDim.x + threadIdx.x;
  int total = Opad * Cpad;
  if (t < total) {
    int o = t / Cpad, c = t - o * Cpad;
    int src = c; bool valid;
    if (split < 0)            { valid = (c < C); }
    else if (c < split)       { valid = true; }
    else if (c < splitAl)     { valid = false; }
    else                      { src = split + (c - splitAl); valid = (src < C); }
    float w = (o < O && valid) ? W[(size_t)o * C + src] : 0.f;
    Wp[(size_t)o * Cpad + c] = (_Float16)w;
  }
  if (t < Opad) {
    float s = 0.f, bb = 0.f;
    if (t < O) {
      s = g ? (g[t] / sqrtf(1.0f + 1e-5f)) : 1.0f;
      float cb  = bcv ? bcv[t] : 0.f;
      float btv = bt ? bt[t] : 0.f;
      bb = cb * s + btv;
    }
    scl[t] = s; bia[t] = bb;
  }
}

__global__ void xyz_mean_k(const float* __restrict__ X, float* __restrict__ mean, int N) {
  __shared__ float sm[3][256];
  int b = blockIdx.x, t = threadIdx.x;
  float a0 = 0, a1 = 0, a2 = 0;
  for (int n = t; n < N; n += 256) {
    const float* p = X + ((size_t)b * N + n) * 3;
    a0 += p[0]; a1 += p[1]; a2 += p[2];
  }
  sm[0][t] = a0; sm[1][t] = a1; sm[2][t] = a2;
  __syncthreads();
  for (int s = 128; s > 0; s >>= 1) {
    if (t < s) { sm[0][t] += sm[0][t + s]; sm[1][t] += sm[1][t + s]; sm[2][t] += sm[2][t + s]; }
    __syncthreads();
  }
  if (t == 0) {
    float inv = 1.0f / (float)N;
    mean[b * 3 + 0] = sm[0][0] * inv;
    mean[b * 3 + 1] = sm[1][0] * inv;
    mean[b * 3 + 2] = sm[2][0] * inv;
  }
}

__global__ void spfe_build_k(const float* __restrict__ X, const float* __restrict__ mean,
                             _Float16* __restrict__ out, int N, int B) {
  int t = blockIdx.x * blockDim.x + threadIdx.x;
  if (t >= B * N) return;
  int b = t / N, n = t - b * N;
  const float* p  = X + ((size_t)b * N + n) * 3;
  const float* mu = mean + b * 3;
  _Float16* row = out + ((size_t)b * N + n) * 32;
  row[0] = (_Float16)p[0]; row[1] = (_Float16)p[1]; row[2] = (_Float16)p[2];
  row[3] = (_Float16)(p[0] - mu[0]); row[4] = (_Float16)(p[1] - mu[1]); row[5] = (_Float16)(p[2] - mu[2]);
  for (int q = 6; q < 32; q++) row[q] = (_Float16)0.f;  // normals=0 + pad
}

__global__ void centers_k(const float* __restrict__ src, float* __restrict__ dst,
                          int Nsrc, int M, int B) {
  int t = blockIdx.x * blockDim.x + threadIdx.x;
  if (t >= B * M) return;
  int b = t / M, m = t - b * M;
  float step = (float)(Nsrc - 1) / (float)(M - 1);
  int idx = (int)((float)m * step);
  if (idx > Nsrc - 1) idx = Nsrc - 1;
  const float* s = src + ((size_t)b * Nsrc + idx) * 3;
  float* d = dst + ((size_t)b * M + m) * 3;
  d[0] = s[0]; d[1] = s[1]; d[2] = s[2];
}

__global__ void knn_k(const float* __restrict__ ctr, const float* __restrict__ pts,
                      int* __restrict__ knn, int M, int Nsrc, int B) {
  int t = blockIdx.x * blockDim.x + threadIdx.x;
  if (t >= B * M) return;
  int b = t / M, m = t - b * M;
  const float* c = ctr + ((size_t)b * M + m) * 3;
  float cx = c[0], cy = c[1], cz = c[2];
  float cn = cx * cx + cy * cy + cz * cz;
  float bd[32]; int bi[32];
#pragma unroll
  for (int i = 0; i < 32; i++) { bd[i] = 3.4e38f; bi[i] = 0; }
  const float* S = pts + (size_t)b * Nsrc * 3;
  for (int n = 0; n < Nsrc; n++) {
    float x = S[n * 3], y = S[n * 3 + 1], z = S[n * 3 + 2];
    float d2 = cn + (x * x + y * y + z * z) - 2.f * (cx * x + cy * y + cz * z);
    if (d2 < bd[31]) {
      int p = 31;
      while (p > 0 && bd[p - 1] > d2) { bd[p] = bd[p - 1]; bi[p] = bi[p - 1]; p--; }
      bd[p] = d2; bi[p] = n;
    }
  }
  int* o = knn + ((size_t)b * M + m) * 32;
  for (int i = 0; i < 32; i++) o[i] = bi[i];
}

// build cat (f16, padded) and the cat-portion + zero pads of the alpha input (single batch)
__global__ void sa_gather_k(const float* __restrict__ xyzS, const float* __restrict__ ctr,
                            const int* __restrict__ knn,
                            const float* __restrict__ feat, int Cfeat, int useXyz,
                            _Float16* __restrict__ catF, int CpadF,
                            _Float16* __restrict__ alphaB, int CpadA, int CcatAl, int Cf,
                            int M) {
  int j = blockIdx.x * blockDim.x + threadIdx.x;
  if (j >= M * 32) return;
  int m = j >> 5, kk = j & 31;
  int n = knn[m * 32 + kk];
  const float* cc = ctr + (size_t)m * 3;
  const float* pp = xyzS + (size_t)n * 3;
  _Float16* crow = catF + (size_t)j * CpadF;
  _Float16* arow = alphaB + (size_t)j * CpadA;
  int c = 0;
  for (int d = 0; d < 3; d++) {
    _Float16 h = (_Float16)(pp[d] - cc[d]);
    crow[c] = h; arow[c] = h; c++;
  }
  if (useXyz) {
    for (int d = 0; d < 3; d++) { _Float16 h = (_Float16)pp[d]; crow[c] = h; arow[c] = h; c++; }
  }
  const float* f = feat + (size_t)n * Cfeat;
  for (int q = 0; q < Cfeat; q++, c++) { _Float16 h = (_Float16)f[q]; crow[c] = h; arow[c] = h; }
  int Ccat = c;
  for (; c < CpadF; c++) crow[c] = (_Float16)0.f;
  for (c = Ccat; c < CcatAl; c++) arow[c] = (_Float16)0.f;
  for (c = CcatAl + Cf; c < CpadA; c++) arow[c] = (_Float16)0.f;
}

// 3-NN inverse-distance interpolation + concat -> f16 GEMM input
__global__ void fp_gather_k(const float* __restrict__ xyzL, const float* __restrict__ xyzH,
                            const float* __restrict__ featH, int Ch,
                            const float* __restrict__ featL, int Cl,
                            _Float16* __restrict__ fcat, int Cpad, int Nl, int Nh, int B) {
  int t = blockIdx.x * blockDim.x + threadIdx.x;
  if (t >= B * Nl) return;
  int b = t / Nl, n = t - b * Nl;
  const float* p = xyzL + ((size_t)b * Nl + n) * 3;
  float px = p[0], py = p[1], pz = p[2];
  float pn = px * px + py * py + pz * pz;
  const float* Hx = xyzH + (size_t)b * Nh * 3;
  float d0 = 3.4e38f, d1 = 3.4e38f, d2 = 3.4e38f;
  int i0 = 0, i1 = 0, i2 = 0;
  for (int h = 0; h < Nh; h++) {
    float x = Hx[h * 3], y = Hx[h * 3 + 1], z = Hx[h * 3 + 2];
    float dd = pn + (x * x + y * y + z * z) - 2.f * (px * x + py * y + pz * z);
    if (dd < d0)      { d2 = d1; i2 = i1; d1 = d0; i1 = i0; d0 = dd; i0 = h; }
    else if (dd < d1) { d2 = d1; i2 = i1; d1 = dd; i1 = h; }
    else if (dd < d2) { d2 = dd; i2 = h; }
  }
  float w0 = 1.f / fmaxf(sqrtf(fmaxf(d0, 1e-12f)), 1e-8f);
  float w1 = 1.f / fmaxf(sqrtf(fmaxf(d1, 1e-12f)), 1e-8f);
  float w2 = 1.f / fmaxf(sqrtf(fmaxf(d2, 1e-12f)), 1e-8f);
  float iw = 1.f / (w0 + w1 + w2);
  w0 *= iw; w1 *= iw; w2 *= iw;
  const float* Fh = featH + (size_t)b * Nh * Ch;
  const float* fa = Fh + (size_t)i0 * Ch;
  const float* fb = Fh + (size_t)i1 * Ch;
  const float* fc = Fh + (size_t)i2 * Ch;
  _Float16* row = fcat + ((size_t)b * Nl + n) * Cpad;
  for (int q = 0; q < Ch; q++) row[q] = (_Float16)(w0 * fa[q] + w1 * fb[q] + w2 * fc[q]);
  const float* Fl = featL + ((size_t)b * Nl + n) * Cl;
  for (int q = 0; q < Cl; q++) row[Ch + q] = (_Float16)Fl[q];
  for (int q = Ch + Cl; q < Cpad; q++) row[q] = (_Float16)0.f;
}

extern "C" void kernel_launch(void* const* d_in, const int* in_sizes, int n_in,
                              void* d_out, int out_size, void* d_ws, size_t ws_size,
                              hipStream_t stream) {
  (void)in_sizes; (void)n_in; (void)out_size; (void)ws_size;
  const int B = 8, N = 4096;
  const float* X = (const float*)d_in[0];

  struct LDef { int iW, iB, iG, iBt, O, C, split; };
  const LDef ld[12] = {
    {1, 2, 3, 4,      64,   9, -1},   // 0 spfe
    {5, 6, 7, 8,     128,  67, -1},   // 1 sa1f
    {9, 10, 11, 12,  128, 195, 67},   // 2 sa1a
    {13, 14, 15, 16, 256, 134, -1},   // 3 sa2f
    {17, 18, 19, 20, 256, 390, 134},  // 4 sa2a
    {21, 22, 23, 24, 512, 262, -1},   // 5 sa3f
    {25, 26, 27, 28, 512, 774, 262},  // 6 sa3a
    {29, 30, 31, 32, 256, 768, -1},   // 7 fp3
    {33, 34, 35, 36, 128, 384, -1},   // 8 fp2
    {37, 38, 39, 40, 128, 192, -1},   // 9 fp1
    {41, 42, 43, 44, 128, 128, -1},   // 10 head1
    {45, 46, -1, -1,  26, 128, -1},   // 11 head2
  };
  int Opad[12], Cpad[12], splitAl[12];
  size_t wOff[12], sOff[12], bOff[12];
  size_t off = 0;
  auto alloc = [&](size_t bytes) { size_t o = off; off += (bytes + 255) & ~(size_t)255; return o; };
  for (int i = 0; i < 12; i++) {
    Opad[i] = (ld[i].O + 15) & ~15;
    int sa = (ld[i].split < 0) ? -1 : ((ld[i].split + 7) & ~7);
    splitAl[i] = sa;
    int effC = (ld[i].split < 0) ? ld[i].C : (sa + (ld[i].C - ld[i].split));
    Cpad[i] = (effC + 31) & ~31;
    wOff[i] = alloc((size_t)Opad[i] * Cpad[i] * 2);
    sOff[i] = alloc((size_t)Opad[i] * 4);
    bOff[i] = alloc((size_t)Opad[i] * 4);
  }
  size_t meanOff    = alloc((size_t)B * 3 * 4);
  size_t spfeCatOff = alloc((size_t)B * N * 32 * 2);
  size_t f0Off      = alloc((size_t)B * N * 64 * 4);
  size_t xyz1Off    = alloc((size_t)B * 1024 * 3 * 4);
  size_t xyz2Off    = alloc((size_t)B * 512 * 3 * 4);
  size_t xyz3Off    = alloc((size_t)B * 256 * 3 * 4);
  size_t knnOff     = alloc((size_t)B * 1024 * 32 * 4);
  size_t f1Off      = alloc((size_t)B * 1024 * 128 * 4);
  size_t f2Off      = alloc((size_t)B * 512 * 256 * 4);
  size_t f3Off      = alloc((size_t)B * 256 * 512 * 4);
  size_t f2uOff     = alloc((size_t)B * 512 * 256 * 4);
  size_t f1uOff     = alloc((size_t)B * 1024 * 128 * 4);
  size_t f0uHOff    = alloc((size_t)B * N * 128 * 2);
  size_t hHOff      = alloc((size_t)B * N * 128 * 2);
  size_t fcat3Off   = alloc((size_t)B * 512 * 768 * 2);
  size_t fcat2Off   = alloc((size_t)B * 1024 * 384 * 2);
  size_t fcat1Off   = alloc((size_t)B * N * 192 * 2);
  // single-batch SA scratch (max over levels)
  size_t catFOff    = alloc((size_t)32768 * 96 * 2);
  size_t alphaOff   = alloc((size_t)32768 * 224 * 2);
  size_t fprOff     = alloc((size_t)32768 * 128 * 2);

  char* W8 = (char*)d_ws;
  auto Fp = [&](size_t o) { return (float*)(W8 + o); };
  auto Hh = [&](size_t o) { return (_Float16*)(W8 + o); };

  // ---- pack weights + fold BN
  for (int i = 0; i < 12; i++) {
    const float* Wq  = (const float*)d_in[ld[i].iW];
    const float* bq  = (const float*)d_in[ld[i].iB];
    const float* gq  = (ld[i].iG  >= 0) ? (const float*)d_in[ld[i].iG]  : nullptr;
    const float* btq = (ld[i].iBt >= 0) ? (const float*)d_in[ld[i].iBt] : nullptr;
    int total = Opad[i] * Cpad[i];
    pack_k<<<(total + 255) / 256, 256, 0, stream>>>(Wq, bq, gq, btq,
        Hh(wOff[i]), Fp(sOff[i]), Fp(bOff[i]),
        ld[i].O, ld[i].C, Opad[i], Cpad[i], ld[i].split, splitAl[i]);
  }

  // ---- SPFE
  xyz_mean_k<<<B, 256, 0, stream>>>(X, Fp(meanOff), N);
  spfe_build_k<<<(B * N + 255) / 256, 256, 0, stream>>>(X, Fp(meanOff), Hh(spfeCatOff), N, B);
  gemm_wmma_k<EPI_BNRELU><<<dim3(N / 128, Opad[0] / 16, B), 128, 0, stream>>>(
      Hh(wOff[0]), Fp(sOff[0]), Fp(bOff[0]),
      Hh(spfeCatOff), (long long)N * 32,
      Fp(f0Off), (long long)N * 64, nullptr, 0,
      nullptr, nullptr, nullptr, 0, 0, N, Cpad[0], 64, 64);

  // ---- SA levels (per-batch to bound scratch)
  auto run_sa = [&](const float* xyzSrc, int Nsrc, float* xyzDst, int M,
                    const float* featPM, int Cfeat, int useXyz,
                    int lf, int la, float* fOut) {
    int Cf = ld[lf].O;
    int Nj = M * 32;
    centers_k<<<(B * M + 255) / 256, 256, 0, stream>>>(xyzSrc, xyzDst, Nsrc, M, B);
    knn_k<<<(B * M + 127) / 128, 128, 0, stream>>>(xyzDst, xyzSrc, (int*)(W8 + knnOff), M, Nsrc, B);
    for (int b = 0; b < B; b++) {
      sa_gather_k<<<(Nj + 127) / 128, 128, 0, stream>>>(
          xyzSrc + (size_t)b * Nsrc * 3, xyzDst + (size_t)b * M * 3,
          (const int*)(W8 + knnOff) + (size_t)b * M * 32,
          featPM + (size_t)b * Nsrc * Cfeat, Cfeat, useXyz,
          Hh(catFOff), Cpad[lf],
          Hh(alphaOff), Cpad[la], splitAl[la], Cf, M);
      gemm_wmma_k<EPI_FPRIME><<<dim3(Nj / 128, Opad[lf] / 16, 1), 128, 0, stream>>>(
          Hh(wOff[lf]), Fp(sOff[lf]), Fp(bOff[lf]),
          Hh(catFOff), 0, nullptr, 0, nullptr, 0,
          nullptr, Hh(fprOff), Hh(alphaOff), Cpad[la], splitAl[la],
          Nj, Cpad[lf], Cf, Cf);
      gemm_wmma_k<EPI_ALPHA><<<dim3(Nj / 128, Opad[la] / 16, 1), 128, 0, stream>>>(
          Hh(wOff[la]), Fp(sOff[la]), Fp(bOff[la]),
          Hh(alphaOff), 0,
          fOut + (size_t)b * M * Cf, 0, nullptr, 0,
          Hh(fprOff), nullptr, nullptr, 0, 0,
          Nj, Cpad[la], Cf, Cf);
    }
  };
  run_sa(X,            N,    Fp(xyz1Off), 1024, Fp(f0Off), 64,  0, 1, 2, Fp(f1Off));
  run_sa(Fp(xyz1Off), 1024,  Fp(xyz2Off), 512,  Fp(f1Off), 128, 1, 3, 4, Fp(f2Off));
  run_sa(Fp(xyz2Off), 512,   Fp(xyz3Off), 256,  Fp(f2Off), 256, 1, 5, 6, Fp(f3Off));

  // ---- FP stages
  auto run_fp = [&](const float* xyzL, int Nl, const float* xyzH, int Nh,
                    const float* featH, int Ch, const float* featL, int Cl,
                    int li, size_t fcatOff, float* outF, _Float16* outH) {
    int Cp = Cpad[li];
    fp_gather_k<<<(B * Nl + 127) / 128, 128, 0, stream>>>(
        xyzL, xyzH, featH, Ch, featL, Cl, Hh(fcatOff), Cp, Nl, Nh, B);
    gemm_wmma_k<EPI_BNRELU><<<dim3(Nl / 128, Opad[li] / 16, B), 128, 0, stream>>>(
        Hh(wOff[li]), Fp(sOff[li]), Fp(bOff[li]),
        Hh(fcatOff), (long long)Nl * Cp,
        outF, outF ? (long long)Nl * ld[li].O : 0,
        outH, outH ? (long long)Nl * ld[li].O : 0,
        nullptr, nullptr, nullptr, 0, 0, Nl, Cp, ld[li].O, ld[li].O);
  };
  run_fp(Fp(xyz2Off), 512,  Fp(xyz3Off), 256,  Fp(f3Off),  512, Fp(f2Off), 256, 7, fcat3Off, Fp(f2uOff), nullptr);
  run_fp(Fp(xyz1Off), 1024, Fp(xyz2Off), 512,  Fp(f2uOff), 256, Fp(f1Off), 128, 8, fcat2Off, Fp(f1uOff), nullptr);
  run_fp(X,           4096, Fp(xyz1Off), 1024, Fp(f1uOff), 128, Fp(f0Off), 64,  9, fcat1Off, nullptr, Hh(f0uHOff));

  // ---- head
  gemm_wmma_k<EPI_BNRELU><<<dim3(N / 128, Opad[10] / 16, B), 128, 0, stream>>>(
      Hh(wOff[10]), Fp(sOff[10]), Fp(bOff[10]),
      Hh(f0uHOff), (long long)N * 128,
      nullptr, 0, Hh(hHOff), (long long)N * 128,
      nullptr, nullptr, nullptr, 0, 0, N, 128, 128, 128);
  gemm_wmma_k<EPI_BIAS><<<dim3(N / 128, Opad[11] / 16, B), 128, 0, stream>>>(
      Hh(wOff[11]), Fp(sOff[11]), Fp(bOff[11]),
      Hh(hHOff), (long long)N * 128,
      (float*)d_out, (long long)26 * N,
      nullptr, 0, nullptr, nullptr, nullptr, 0, 0, N, 128, 26, 26);
}
